// EngagementBiasedMHA_74088185856182
// MI455X (gfx1250) — compile-verified, matrix operand order checked by reading, not run on hardware
//
#include <hip/hip_runtime.h>
#include <hip/hip_bf16.h>

typedef __bf16 bf16;
typedef __attribute__((ext_vector_type(16))) __bf16 v16bf;
typedef __attribute__((ext_vector_type(8)))  __bf16 v8bf;
typedef __attribute__((ext_vector_type(8)))  float  v8f;

// Problem constants
constexpr int B_  = 4;
constexpr int T_  = 2048;
constexpr int D_  = 1024;
constexpr int H_  = 16;
constexpr int HD_ = 64;
constexpr int NKB = T_ / 64;   // key blocks in attention

// ---------------------------------------------------------------------------
// WMMA + async helpers (CDNA5 gfx1250, wave32)
// ---------------------------------------------------------------------------
__device__ __forceinline__ v8f wmma_bf16(v16bf a, v16bf b, v8f c) {
  return __builtin_amdgcn_wmma_f32_16x16x32_bf16(false, a, false, b, (short)0, c,
                                                 false, false);
}

// Async DMA: copy 16 bytes global -> LDS, tracked by ASYNCcnt. LDS operand is
// the wave-relative LDS byte address (low 32 bits of the generic pointer).
__device__ __forceinline__ void async_copy_b128(const bf16* src, bf16* lds_dst) {
#if defined(__gfx1250__)
  unsigned lds_addr = (unsigned)(unsigned long long)lds_dst;
  unsigned long long gaddr = (unsigned long long)src;
  asm volatile("global_load_async_to_lds_b128 %0, %1, off"
               :: "v"(lds_addr), "v"(gaddr) : "memory");
#else
  *(v8bf*)lds_dst = *(const v8bf*)src;
#endif
}

__device__ __forceinline__ void async_wait0() {
#if defined(__gfx1250__)
  asm volatile("s_wait_asynccnt 0" ::: "memory");
#endif
}

// Load a 16x32 bf16 A-fragment (or B-fragment of X*W^T, W row-major [N,K]).
// p points at element [row0, k0]; ld in elements.  Per ISA 7.12.2.
__device__ __forceinline__ v16bf load_frag(const bf16* __restrict__ p, int ld) {
  const int lane = threadIdx.x & 31;
  const int hl   = lane >> 4;
  const int r    = lane & 15;
  const bf16* base = p + (size_t)r * ld + 8 * hl;
  v8bf lo = *(const v8bf*)(base);
  v8bf hi = *(const v8bf*)(base + 16);
  return __builtin_shufflevector(lo, hi, 0,1,2,3,4,5,6,7,8,9,10,11,12,13,14,15);
}

// ---------------------------------------------------------------------------
// Kernel 1: fp32 -> bf16 conversion + per-key log-engagement/mask bias
// ---------------------------------------------------------------------------
__global__ void prep_kernel(const float* __restrict__ x,
                            const float* __restrict__ qkv_w,
                            const float* __restrict__ out_w,
                            const float* __restrict__ eng,
                            const unsigned char* __restrict__ mask,
                            bf16* __restrict__ xb,
                            bf16* __restrict__ w1b,
                            bf16* __restrict__ w2b,
                            float* __restrict__ biasb) {
  const long i = (long)blockIdx.x * blockDim.x + threadIdx.x;
  const long stride = (long)gridDim.x * blockDim.x;
  const long NX  = (long)B_ * T_ * D_ / 8;
  const long NW1 = (long)3 * D_ * D_ / 8;
  const long NW2 = (long)D_ * D_ / 8;
  const v8f* xs  = (const v8f*)x;   v8bf* xd  = (v8bf*)xb;
  const v8f* w1s = (const v8f*)qkv_w; v8bf* w1d = (v8bf*)w1b;
  const v8f* w2s = (const v8f*)out_w; v8bf* w2d = (v8bf*)w2b;
  for (long j = i; j < NX;  j += stride) xd[j]  = __builtin_convertvector(xs[j], v8bf);
  for (long j = i; j < NW1; j += stride) w1d[j] = __builtin_convertvector(w1s[j], v8bf);
  for (long j = i; j < NW2; j += stride) w2d[j] = __builtin_convertvector(w2s[j], v8bf);
  for (long j = i; j < (long)B_ * T_; j += stride) {
    float e = fmaxf(eng[j], 1e-6f);
    biasb[j] = mask[j] ? -1e9f : __logf(e);
  }
}

// ---------------------------------------------------------------------------
// Kernel 2: QKV projection, software-pipelined.  Wave tile 32(M) x 64(N),
// 8 waves/block => block tile 64 x 256.  Q,K head-major [B,H,T,HD];
// V stored TRANSPOSED per head: [B,H,HD,T]  (so attention can DMA V^T tiles).
// ---------------------------------------------------------------------------
__global__ __launch_bounds__(256)
void qkv_gemm_kernel(const bf16* __restrict__ A, const bf16* __restrict__ Wt,
                     const float* __restrict__ qb,
                     bf16* __restrict__ Qo, bf16* __restrict__ Ko,
                     bf16* __restrict__ Vo) {
  const int tid  = threadIdx.x;
  const int lane = tid & 31;
  const int wave = tid >> 5;
  const int hl   = lane >> 4;
  const int lx   = lane & 15;
  const int wm   = wave & 1;
  const int wn   = wave >> 1;
  const int m0 = blockIdx.x * 64 + wm * 32;
  const int n0 = blockIdx.y * 256 + wn * 64;

  const bf16* A0 = A + (size_t)m0 * D_;
  const bf16* A1 = A + (size_t)(m0 + 16) * D_;
  const bf16* W0 = Wt + (size_t)(n0 +  0) * D_;
  const bf16* W1 = Wt + (size_t)(n0 + 16) * D_;
  const bf16* W2 = Wt + (size_t)(n0 + 32) * D_;
  const bf16* W3 = Wt + (size_t)(n0 + 48) * D_;

  v16bf a0c = load_frag(A0, D_), a1c = load_frag(A1, D_);
  v16bf b0c = load_frag(W0, D_), b1c = load_frag(W1, D_);
  v16bf b2c = load_frag(W2, D_), b3c = load_frag(W3, D_);

  v8f c[8] = {};
  for (int k0 = 32; k0 < D_; k0 += 32) {
    // prefetch next K-slice while current WMMAs run
    v16bf a0n = load_frag(A0 + k0, D_);
    v16bf a1n = load_frag(A1 + k0, D_);
    v16bf b0n = load_frag(W0 + k0, D_);
    v16bf b1n = load_frag(W1 + k0, D_);
    v16bf b2n = load_frag(W2 + k0, D_);
    v16bf b3n = load_frag(W3 + k0, D_);
    c[0] = wmma_bf16(a0c, b0c, c[0]);
    c[1] = wmma_bf16(a0c, b1c, c[1]);
    c[2] = wmma_bf16(a0c, b2c, c[2]);
    c[3] = wmma_bf16(a0c, b3c, c[3]);
    c[4] = wmma_bf16(a1c, b0c, c[4]);
    c[5] = wmma_bf16(a1c, b1c, c[5]);
    c[6] = wmma_bf16(a1c, b2c, c[6]);
    c[7] = wmma_bf16(a1c, b3c, c[7]);
    a0c = a0n; a1c = a1n;
    b0c = b0n; b1c = b1n; b2c = b2n; b3c = b3n;
  }
  c[0] = wmma_bf16(a0c, b0c, c[0]);
  c[1] = wmma_bf16(a0c, b1c, c[1]);
  c[2] = wmma_bf16(a0c, b2c, c[2]);
  c[3] = wmma_bf16(a0c, b3c, c[3]);
  c[4] = wmma_bf16(a1c, b0c, c[4]);
  c[5] = wmma_bf16(a1c, b1c, c[5]);
  c[6] = wmma_bf16(a1c, b2c, c[6]);
  c[7] = wmma_bf16(a1c, b3c, c[7]);

#pragma unroll
  for (int g = 0; g < 4; ++g) {
    const int n    = n0 + 16 * g + lx;
    const float bb = qb[n];
    const int sect = n >> 10;          // 0=Q 1=K 2=V
    const int nn   = n & (D_ - 1);
    const int h    = nn >> 6;
    const int hd   = nn & 63;
    bf16* dst = (sect == 0) ? Qo : (sect == 1) ? Ko : Vo;
#pragma unroll
    for (int sm = 0; sm < 2; ++sm) {
      const v8f cc = c[sm * 4 + g];
#pragma unroll
      for (int j = 0; j < 8; ++j) {
        const int m  = m0 + sm * 16 + hl * 8 + j;
        const int bi = m >> 11;
        const int t  = m & (T_ - 1);
        size_t addr;
        if (sect == 2)  // V^T: [B,H,HD,T]
          addr = (((size_t)bi * H_ + h) * HD_ + hd) * T_ + t;
        else            // Q,K: [B,H,T,HD]
          addr = (((size_t)bi * H_ + h) * T_ + t) * HD_ + hd;
        dst[addr] = (bf16)(cc[j] + bb);
      }
    }
  }
}

// ---------------------------------------------------------------------------
// Kernel 3: flash attention per (b,h).  Block = 128 threads (4 waves), each
// wave owns 16 q rows of the 64-row tile.  K and V^T tiles are staged with
// async global->LDS DMA into double-buffered LDS, one key-block ahead.
// ---------------------------------------------------------------------------
__global__ __launch_bounds__(128)
void attn_kernel(const bf16* __restrict__ Q, const bf16* __restrict__ K,
                 const bf16* __restrict__ Vt, const float* __restrict__ bias,
                 bf16* __restrict__ O) {
  __shared__ bf16 sK[2][64 * 64];     // [key][hd] row-major
  __shared__ bf16 sVt[2][64 * 64];    // [hd][key] (already transposed)
  __shared__ bf16 sP[4][16 * 64];     // per-wave probabilities [qrow][key]

  const int tid  = threadIdx.x;
  const int lane = tid & 31;
  const int wave = tid >> 5;
  const int hl   = lane >> 4;
  const int lx   = lane & 15;
  const int bh   = blockIdx.y;
  const int b    = bh >> 4;
  const int h    = bh & 15;
  const int m0   = blockIdx.x * 64 + wave * 16;

  const bf16* gK  = K  + (size_t)bh * T_ * HD_;   // [t][hd]
  const bf16* gVt = Vt + (size_t)bh * HD_ * T_;   // [hd][t]

  // Stage one 64-key block (8KB K + 8KB V^T) into LDS buffer `buf`.
  auto stageK = [&](int kb, int buf) {
    const bf16* src = gK + (size_t)kb * 64 * HD_;  // contiguous 8KB
    bf16* dst = sK[buf];
    for (int i = tid; i < 512; i += 128)
      async_copy_b128(src + i * 8, dst + i * 8);
  };
  auto stageV = [&](int kb, int buf) {
    bf16* dst = sVt[buf];
    for (int i = tid; i < 512; i += 128) {
      const int row = i >> 3;          // hd
      const int ch  = i & 7;           // 16B chunk within the 64-key row
      async_copy_b128(gVt + (size_t)row * T_ + kb * 64 + ch * 8,
                      dst + row * 64 + ch * 8);
    }
  };

  const bf16* qbase = Q + ((size_t)bh * T_ + m0) * HD_;
  const v16bf a0 = load_frag(qbase + 0, HD_);
  const v16bf a1 = load_frag(qbase + 32, HD_);

  v8f o0 = {}, o1 = {}, o2 = {}, o3 = {};
  float rmax[8], rsum[8];
#pragma unroll
  for (int j = 0; j < 8; ++j) { rmax[j] = -3.0e38f; rsum[j] = 0.0f; }

  const float scale = 0.125f;   // 1/sqrt(64)

  // Prologue: stage block 0, then issue block 1 behind it.
  stageK(0, 0); stageV(0, 0);
  async_wait0();
  __syncthreads();
  stageK(1, 1); stageV(1, 1);

  for (int kb = 0; kb < NKB; ++kb) {
    const int cur = kb & 1;
    const bf16* cK = sK[cur];
    const bf16* cV = sVt[cur];

    // S = Q K^T
    v8f s[4];
#pragma unroll
    for (int g = 0; g < 4; ++g) {
      v8f acc = {};
      acc = wmma_bf16(a0, load_frag(cK + (16 * g) * 64 + 0, 64), acc);
      acc = wmma_bf16(a1, load_frag(cK + (16 * g) * 64 + 32, 64), acc);
      s[g] = acc;
    }

    // scale + per-key engagement/mask bias
#pragma unroll
    for (int g = 0; g < 4; ++g) {
      const float bb = bias[b * T_ + kb * 64 + 16 * g + lx];
#pragma unroll
      for (int j = 0; j < 8; ++j) s[g][j] = s[g][j] * scale + bb;
    }

    // online softmax: row max across the 64-key block (16-lane half reduce)
    float nmax[8];
#pragma unroll
    for (int j = 0; j < 8; ++j) {
      float m = fmaxf(fmaxf(s[0][j], s[1][j]), fmaxf(s[2][j], s[3][j]));
      m = fmaxf(m, __shfl_xor(m, 1, 32));
      m = fmaxf(m, __shfl_xor(m, 2, 32));
      m = fmaxf(m, __shfl_xor(m, 4, 32));
      m = fmaxf(m, __shfl_xor(m, 8, 32));
      nmax[j] = fmaxf(rmax[j], m);
    }

    // exp, write P to per-wave LDS, partial row sums
    float ps[8];
#pragma unroll
    for (int j = 0; j < 8; ++j) ps[j] = 0.0f;
    bf16* pw = sP[wave];
#pragma unroll
    for (int g = 0; g < 4; ++g) {
#pragma unroll
      for (int j = 0; j < 8; ++j) {
        const float p = __expf(s[g][j] - nmax[j]);
        ps[j] += p;
        pw[(hl * 8 + j) * 64 + 16 * g + lx] = (bf16)p;
      }
    }

    // rescale accumulators + update running stats
#pragma unroll
    for (int j = 0; j < 8; ++j) {
      const float alpha = __expf(rmax[j] - nmax[j]);
      rmax[j] = nmax[j];
      float t = ps[j];
      t += __shfl_xor(t, 1, 32);
      t += __shfl_xor(t, 2, 32);
      t += __shfl_xor(t, 4, 32);
      t += __shfl_xor(t, 8, 32);
      rsum[j] = rsum[j] * alpha + t;
      o0[j] *= alpha; o1[j] *= alpha; o2[j] *= alpha; o3[j] *= alpha;
    }

    // O += P V  (A = P from per-wave LDS; B = V^T tile; DS in-order per wave)
#pragma unroll
    for (int kk = 0; kk < 2; ++kk) {
      v16bf pa = load_frag(pw + kk * 32, 64);
      o0 = wmma_bf16(pa, load_frag(cV + 0 * 16 * 64 + kk * 32, 64), o0);
      o1 = wmma_bf16(pa, load_frag(cV + 1 * 16 * 64 + kk * 32, 64), o1);
      o2 = wmma_bf16(pa, load_frag(cV + 2 * 16 * 64 + kk * 32, 64), o2);
      o3 = wmma_bf16(pa, load_frag(cV + 3 * 16 * 64 + kk * 32, 64), o3);
    }

    __syncthreads();                 // everyone done reading buffer `cur`
    if (kb + 1 < NKB) {
      async_wait0();                 // block kb+1 fully landed in LDS
      if (kb + 2 < NKB) {            // refill the buffer we just freed
        stageK(kb + 2, cur);
        stageV(kb + 2, cur);
      }
      __syncthreads();               // kb+1 visible to all waves
    }
  }

  // Normalize and write O token-major [B,T,D] as bf16
#pragma unroll
  for (int j = 0; j < 8; ++j) {
    const int trow = m0 + hl * 8 + j;
    const float inv = 1.0f / rsum[j];
    const size_t base = ((size_t)b * T_ + trow) * D_ + h * HD_ + lx;
    O[base + 0]  = (bf16)(o0[j] * inv);
    O[base + 16] = (bf16)(o1[j] * inv);
    O[base + 32] = (bf16)(o2[j] * inv);
    O[base + 48] = (bf16)(o3[j] * inv);
  }
}

// ---------------------------------------------------------------------------
// Kernel 4: output projection (same pipelined scheme), fp32 output.
// ---------------------------------------------------------------------------
__global__ __launch_bounds__(256)
void out_gemm_kernel(const bf16* __restrict__ A, const bf16* __restrict__ Wt,
                     const float* __restrict__ ob, float* __restrict__ out) {
  const int tid  = threadIdx.x;
  const int lane = tid & 31;
  const int wave = tid >> 5;
  const int hl   = lane >> 4;
  const int lx   = lane & 15;
  const int wm   = wave & 1;
  const int wn   = wave >> 1;
  const int m0 = blockIdx.x * 64 + wm * 32;
  const int n0 = blockIdx.y * 256 + wn * 64;

  const bf16* A0 = A + (size_t)m0 * D_;
  const bf16* A1 = A + (size_t)(m0 + 16) * D_;
  const bf16* W0 = Wt + (size_t)(n0 +  0) * D_;
  const bf16* W1 = Wt + (size_t)(n0 + 16) * D_;
  const bf16* W2 = Wt + (size_t)(n0 + 32) * D_;
  const bf16* W3 = Wt + (size_t)(n0 + 48) * D_;

  v16bf a0c = load_frag(A0, D_), a1c = load_frag(A1, D_);
  v16bf b0c = load_frag(W0, D_), b1c = load_frag(W1, D_);
  v16bf b2c = load_frag(W2, D_), b3c = load_frag(W3, D_);

  v8f c[8] = {};
  for (int k0 = 32; k0 < D_; k0 += 32) {
    v16bf a0n = load_frag(A0 + k0, D_);
    v16bf a1n = load_frag(A1 + k0, D_);
    v16bf b0n = load_frag(W0 + k0, D_);
    v16bf b1n = load_frag(W1 + k0, D_);
    v16bf b2n = load_frag(W2 + k0, D_);
    v16bf b3n = load_frag(W3 + k0, D_);
    c[0] = wmma_bf16(a0c, b0c, c[0]);
    c[1] = wmma_bf16(a0c, b1c, c[1]);
    c[2] = wmma_bf16(a0c, b2c, c[2]);
    c[3] = wmma_bf16(a0c, b3c, c[3]);
    c[4] = wmma_bf16(a1c, b0c, c[4]);
    c[5] = wmma_bf16(a1c, b1c, c[5]);
    c[6] = wmma_bf16(a1c, b2c, c[6]);
    c[7] = wmma_bf16(a1c, b3c, c[7]);
    a0c = a0n; a1c = a1n;
    b0c = b0n; b1c = b1n; b2c = b2n; b3c = b3n;
  }
  c[0] = wmma_bf16(a0c, b0c, c[0]);
  c[1] = wmma_bf16(a0c, b1c, c[1]);
  c[2] = wmma_bf16(a0c, b2c, c[2]);
  c[3] = wmma_bf16(a0c, b3c, c[3]);
  c[4] = wmma_bf16(a1c, b0c, c[4]);
  c[5] = wmma_bf16(a1c, b1c, c[5]);
  c[6] = wmma_bf16(a1c, b2c, c[6]);
  c[7] = wmma_bf16(a1c, b3c, c[7]);

#pragma unroll
  for (int g = 0; g < 4; ++g) {
    const int n = n0 + 16 * g + lx;
    const float bb = ob[n];
#pragma unroll
    for (int sm = 0; sm < 2; ++sm) {
      const v8f cc = c[sm * 4 + g];
#pragma unroll
      for (int j = 0; j < 8; ++j) {
        const int m = m0 + sm * 16 + hl * 8 + j;
        out[(size_t)m * D_ + n] = cc[j] + bb;
      }
    }
  }
}

// ---------------------------------------------------------------------------
// Host launcher
// ---------------------------------------------------------------------------
extern "C" void kernel_launch(void* const* d_in, const int* in_sizes, int n_in,
                              void* d_out, int out_size, void* d_ws, size_t ws_size,
                              hipStream_t stream) {
  const float* x      = (const float*)d_in[0];
  const float* eng    = (const float*)d_in[1];
  const unsigned char* mask = (const unsigned char*)d_in[2];
  const float* qkv_w  = (const float*)d_in[3];
  const float* qkv_b  = (const float*)d_in[4];
  const float* out_w  = (const float*)d_in[5];
  const float* out_b  = (const float*)d_in[6];
  float* out = (float*)d_out;

  size_t off = 0;
  auto alloc = [&](size_t bytes) -> void* {
    void* r = (char*)d_ws + off;
    off += (bytes + 255) & ~(size_t)255;
    return r;
  };
  bf16* xb   = (bf16*)alloc((size_t)B_ * T_ * D_ * sizeof(bf16));
  bf16* w1b  = (bf16*)alloc((size_t)3 * D_ * D_ * sizeof(bf16));
  bf16* w2b  = (bf16*)alloc((size_t)D_ * D_ * sizeof(bf16));
  bf16* Qb   = (bf16*)alloc((size_t)B_ * T_ * D_ * sizeof(bf16));
  bf16* Kb   = (bf16*)alloc((size_t)B_ * T_ * D_ * sizeof(bf16));
  bf16* Vtb  = (bf16*)alloc((size_t)B_ * T_ * D_ * sizeof(bf16));
  bf16* Ob   = (bf16*)alloc((size_t)B_ * T_ * D_ * sizeof(bf16));
  float* biasb = (float*)alloc((size_t)B_ * T_ * sizeof(float));

  prep_kernel<<<2048, 256, 0, stream>>>(x, qkv_w, out_w, eng, mask,
                                        xb, w1b, w2b, biasb);

  // M=8192 (128 blocks of 64), N=3072 (12 blocks of 256)
  qkv_gemm_kernel<<<dim3(128, 12), 256, 0, stream>>>(xb, w1b, qkv_b, Qb, Kb, Vtb);

  // 32 q-tiles of 64 rows, B*H = 64 (b,h) pairs
  attn_kernel<<<dim3(32, 64), 128, 0, stream>>>(Qb, Kb, Vtb, biasb, Ob);

  // M=8192, N=1024 (4 blocks of 256)
  out_gemm_kernel<<<dim3(128, 4), 256, 0, stream>>>(Ob, w2b, out_b, out);
}
